// OctreeResblock_15479062134909
// MI455X (gfx1250) — matrix operand order repaired and checked by reading, not compiled, not run on hardware
//
#include <hip/hip_runtime.h>

// ---------------------------------------------------------------------------
// Octree residual block for MI455X (gfx1250, wave32, WMMA).
//
// Strategy: scatter->gather inversion over a dense bf16 feature grid (16 MB,
// L2-resident), v_wmma_f32_16x16x32_bf16 for the per-tap 32x32 channel mixes,
// weights staged in LDS, zero atomics. Each wave computes TWO 16-point tiles
// so every LDS B-fragment read feeds 2 WMMAs (4 WMMAs per tap per wave).
// ---------------------------------------------------------------------------

typedef __attribute__((ext_vector_type(16))) __bf16 v16bf;
typedef __attribute__((ext_vector_type(8)))  float  v8f;

#define OC_N 65536
#define OC_D 64
#define OC_C 32
#define OC_G (OC_D * OC_D * OC_D)       // 262144 cells (B == 1)
#define OC_TAPS 27

union AFrag { uint4 u[2]; v16bf v; };

__device__ __forceinline__ int oc_clamp(int v) {
    return v < 0 ? 0 : (v > OC_D - 1 ? OC_D - 1 : v);
}

// ---------------------------------------------------------------------------
// 1) Zero the two bf16 feature grids (32 MB total).
// ---------------------------------------------------------------------------
__global__ void oc_zero_kernel(uint4* __restrict__ p, int n16) {
    int i = blockIdx.x * blockDim.x + threadIdx.x;
    if (i < n16) p[i] = make_uint4(0u, 0u, 0u, 0u);
}

// ---------------------------------------------------------------------------
// 2) Pack f32 kernel taps (27 x Cin x Cout) into bf16 WMMA B-fragment layout.
//    wpack[((t*2 + h)*32 + lane)*16 + e] = kern[t][ (e<8 ? 0 : 16) + (lane>=16?8:0) + (e&7) ][ h*16 + (lane&15) ]
//    so a lane's 16 elements are contiguous (two 16B LDS loads).
// ---------------------------------------------------------------------------
__global__ void oc_pack_weights_kernel(const float* __restrict__ kern,
                                       __bf16* __restrict__ wpack) {
    int tid = blockIdx.x * blockDim.x + threadIdx.x;   // 27*2*32 = 1728 threads
    if (tid >= OC_TAPS * 2 * 32) return;
    int t    = tid >> 6;
    int rest = tid & 63;
    int h    = rest >> 5;        // output-column half: 0 -> cols 0..15, 1 -> cols 16..31
    int lane = rest & 31;
    int kbase = (lane >= 16) ? 8 : 0;
    int col   = h * 16 + (lane & 15);
    __bf16* dst = wpack + ((size_t)((t * 2 + h) * 32 + lane)) * 16;
    #pragma unroll
    for (int e = 0; e < 16; ++e) {
        int cin = (e < 8) ? kbase + e : 16 + kbase + (e - 8);
        dst[e] = (__bf16)kern[(t * OC_C + cin) * OC_C + col];
    }
}

// ---------------------------------------------------------------------------
// 3) Scatter input point features into the dense bf16 grid.
//    Point coordinates are unique -> plain stores, no atomics.
// ---------------------------------------------------------------------------
__global__ void oc_scatter_values_kernel(const float* __restrict__ values,
                                         const long long* __restrict__ indices,
                                         __bf16* __restrict__ grid) {
    int tid = blockIdx.x * blockDim.x + threadIdx.x;   // N * 32 threads
    if (tid >= OC_N * OC_C) return;
    int n = tid >> 5;
    int c = tid & 31;
    int z = (int)indices[n * 4 + 1];
    int y = (int)indices[n * 4 + 2];
    int x = (int)indices[n * 4 + 3];
    int lin = (z * OC_D + y) * OC_D + x;
    grid[(size_t)lin * OC_C + c] = (__bf16)values[n * OC_C + c];
}

// ---------------------------------------------------------------------------
// 4) Gather conv: one wave computes 32 points x 32 output channels as two
//    16-point WMMA tiles sharing the B fragments. Per tap: 2 bf16 A gathers
//    (16x32 each), 2 B fragments (32x16) from LDS, 4x v_wmma_f32_16x16x32_bf16.
// ---------------------------------------------------------------------------
template <bool RELU, bool FINAL>
__global__ void __launch_bounds__(128)
oc_conv_kernel(const __bf16* __restrict__ grid_in,
               const long long* __restrict__ indices,
               const __bf16* __restrict__ wpack,
               const float* __restrict__ bias,
               const float* __restrict__ mask,
               const float* __restrict__ residual,   // FINAL only
               __bf16* __restrict__ grid_out,        // !FINAL only
               float* __restrict__ out)              // FINAL only
{
    __shared__ __bf16 wlds[OC_TAPS * 1024];          // 55296 B of packed weights
    __shared__ int    lins_s[4 * 32];

    const int tid = threadIdx.x;

    // Cooperative weight load into LDS: 27*1024 bf16 = 3456 uint4 / 128 threads.
    {
        const uint4* ws = reinterpret_cast<const uint4*>(wpack);
        uint4*       wd = reinterpret_cast<uint4*>(wlds);
        #pragma unroll
        for (int i = 0; i < 27; ++i) wd[i * 128 + tid] = ws[i * 128 + tid];
    }

    const int wave = tid >> 5;
    const int lane = tid & 31;
    const int row  = lane & 15;
    const int half = lane >> 4;                        // 0 or 1 (K-offset half)
    const int p0   = (blockIdx.x * 4 + wave) * 32;     // first point of this wave
    const int n0   = p0 + row;                         // tile 0 row
    const int n1   = p0 + 16 + row;                    // tile 1 row

    const int z0 = (int)indices[n0 * 4 + 1];
    const int y0 = (int)indices[n0 * 4 + 2];
    const int x0 = (int)indices[n0 * 4 + 3];
    const int z1 = (int)indices[n1 * 4 + 1];
    const int y1 = (int)indices[n1 * 4 + 2];
    const int x1 = (int)indices[n1 * 4 + 3];
    if (half == 0) {
        lins_s[wave * 32 + row]      = (z0 * OC_D + y0) * OC_D + x0;
        lins_s[wave * 32 + 16 + row] = (z1 * OC_D + y1) * OC_D + x1;
    }
    __syncthreads();

    v8f acc00 = {}, acc01 = {};   // tile 0: cols 0..15 / 16..31
    v8f acc10 = {}, acc11 = {};   // tile 1: cols 0..15 / 16..31
    const int kb = half * 8;      // A-fragment K base per lane half

    #pragma unroll 3
    for (int t = 0; t < OC_TAPS; ++t) {
        const int oz = t / 9 - 1;
        const int oy = (t / 3) % 3 - 1;
        const int ox = t % 3 - 1;
        // Gather sources (inverse of the reference's scatter offset), clamped.
        const int slin0 = (oc_clamp(z0 - oz) * OC_D + oc_clamp(y0 - oy)) * OC_D + oc_clamp(x0 - ox);
        const int slin1 = (oc_clamp(z1 - oz) * OC_D + oc_clamp(y1 - oy)) * OC_D + oc_clamp(x1 - ox);
        const __bf16* cell0 = grid_in + (size_t)slin0 * OC_C;
        const __bf16* cell1 = grid_in + (size_t)slin1 * OC_C;

        AFrag a0, a1;
        a0.u[0] = *reinterpret_cast<const uint4*>(cell0 + kb);        // K kb..kb+7
        a0.u[1] = *reinterpret_cast<const uint4*>(cell0 + kb + 16);   // K kb+16..kb+23
        a1.u[0] = *reinterpret_cast<const uint4*>(cell1 + kb);
        a1.u[1] = *reinterpret_cast<const uint4*>(cell1 + kb + 16);

        AFrag b0, b1;
        const uint4* wl0 = reinterpret_cast<const uint4*>(wlds + ((t * 2 + 0) * 32 + lane) * 16);
        const uint4* wl1 = reinterpret_cast<const uint4*>(wlds + ((t * 2 + 1) * 32 + lane) * 16);
        b0.u[0] = wl0[0]; b0.u[1] = wl0[1];
        b1.u[0] = wl1[0]; b1.u[1] = wl1[1];

        acc00 = __builtin_amdgcn_wmma_f32_16x16x32_bf16(false, a0.v, false, b0.v,
                                                        (short)0, acc00, false, false);
        acc01 = __builtin_amdgcn_wmma_f32_16x16x32_bf16(false, a0.v, false, b1.v,
                                                        (short)0, acc01, false, false);
        acc10 = __builtin_amdgcn_wmma_f32_16x16x32_bf16(false, a1.v, false, b0.v,
                                                        (short)0, acc10, false, false);
        acc11 = __builtin_amdgcn_wmma_f32_16x16x32_bf16(false, a1.v, false, b1.v,
                                                        (short)0, acc11, false, false);
    }

    // Writeback. C/D layout: element r of acc holds row (r + half*8), col = lane&15.
    const int col = row;
    const float bv0 = bias[col];
    const float bv1 = bias[col + 16];
    #pragma unroll
    for (int tile = 0; tile < 2; ++tile) {
        const v8f& a_lo = tile ? acc10 : acc00;
        const v8f& a_hi = tile ? acc11 : acc01;
        #pragma unroll
        for (int r = 0; r < 8; ++r) {
            const int mrow = tile * 16 + r + half * 8;
            const int m    = p0 + mrow;
            const float mk = mask[m];
            float o0 = (a_lo[r] + mk * bv0) * mk;
            float o1 = (a_hi[r] + mk * bv1) * mk;
            if (RELU) { o0 = fmaxf(o0, 0.0f); o1 = fmaxf(o1, 0.0f); }
            if (FINAL) {
                out[(size_t)m * OC_C + col]      = residual[(size_t)m * OC_C + col]      + o0;
                out[(size_t)m * OC_C + col + 16] = residual[(size_t)m * OC_C + col + 16] + o1;
            } else {
                const int mlin = lins_s[wave * 32 + mrow];
                grid_out[(size_t)mlin * OC_C + col]      = (__bf16)o0;
                grid_out[(size_t)mlin * OC_C + col + 16] = (__bf16)o1;
            }
        }
    }
}

// ---------------------------------------------------------------------------
// Launch: zero grids -> pack weights -> scatter values -> conv1 -> conv2.
// ---------------------------------------------------------------------------
extern "C" void kernel_launch(void* const* d_in, const int* in_sizes, int n_in,
                              void* d_out, int out_size, void* d_ws, size_t ws_size,
                              hipStream_t stream) {
    (void)in_sizes; (void)n_in; (void)out_size; (void)ws_size;

    const float*     values = (const float*)d_in[0];
    const long long* indices = (const long long*)d_in[1];   // reference: int64
    const float*     mask   = (const float*)d_in[2];
    const float*     kern1  = (const float*)d_in[3];
    const float*     bias1  = (const float*)d_in[4];
    const float*     kern2  = (const float*)d_in[5];
    const float*     bias2  = (const float*)d_in[6];
    float*           out    = (float*)d_out;

    char* base = (char*)d_ws;
    const size_t GRID_BYTES = (size_t)OC_G * OC_C * sizeof(__bf16);   // 16 MB
    __bf16* grid1 = (__bf16*)base;
    __bf16* grid2 = (__bf16*)(base + GRID_BYTES);
    __bf16* w1    = (__bf16*)(base + 2 * GRID_BYTES);
    __bf16* w2    = (__bf16*)(base + 2 * GRID_BYTES + 64 * 1024);

    // 1) Zero both grids (2 x 16 MB) as uint4.
    {
        int n16 = (int)(2 * GRID_BYTES / 16);
        oc_zero_kernel<<<(n16 + 255) / 256, 256, 0, stream>>>((uint4*)base, n16);
    }
    // 2) Pack weights to bf16 B-fragment layout.
    {
        int nt = OC_TAPS * 2 * 32;
        oc_pack_weights_kernel<<<(nt + 127) / 128, 128, 0, stream>>>(kern1, w1);
        oc_pack_weights_kernel<<<(nt + 127) / 128, 128, 0, stream>>>(kern2, w2);
    }
    // 3) Scatter input features into grid1 (unique cells -> no atomics).
    {
        int nt = OC_N * OC_C;
        oc_scatter_values_kernel<<<(nt + 255) / 256, 256, 0, stream>>>(values, indices, grid1);
    }
    // 4) Conv1 (bias + ReLU) -> grid2 ; Conv2 (bias) + residual -> out.
    const int conv_blocks = OC_N / 128;   // 4 waves/block, 32 points/wave
    oc_conv_kernel<true,  false><<<conv_blocks, 128, 0, stream>>>(
        grid1, indices, w1, bias1, mask, nullptr, grid2, nullptr);
    oc_conv_kernel<false, true ><<<conv_blocks, 128, 0, stream>>>(
        grid2, indices, w2, bias2, mask, values, nullptr, out);
}
